// CasualMutiHeadSelfAttention_60593398612506
// MI455X (gfx1250) — matrix operand order repaired and checked by reading, not compiled, not run on hardware
//
#include <hip/hip_runtime.h>
#include <hip/hip_bf16.h>
#include <math.h>

// Causal MHA fwd for MI455X (gfx1250, wave32, WMMA, TDM).
// Compute-bound (~137 GFLOP vs ~150MB traffic @23.3TB/s) -> bf16 WMMA with f32 accum.
// Attention streams K/V tiles into LDS with the Tensor Data Mover (TENSORcnt
// double-buffering) so the WMMA pipe overlaps with DMA instead of VMEM latency.

#define BB  4
#define SS  2048
#define DD  1024
#define HH  16
#define DKK 64
#define BSR (BB * SS) // 8192 rows
#define AW  4         // waves per block in attention kernel

typedef __bf16 bf16;
typedef __attribute__((ext_vector_type(16))) __bf16        v16bf;
typedef __attribute__((ext_vector_type(8)))  float         v8f;
typedef __attribute__((ext_vector_type(4)))  unsigned int  v4u;
typedef __attribute__((ext_vector_type(8)))  int           v8i;
typedef __attribute__((ext_vector_type(4)))  int           v4i;

#ifndef __has_builtin
#define __has_builtin(x) 0
#endif
#if defined(__HIP_DEVICE_COMPILE__) && __has_builtin(__builtin_amdgcn_tensor_load_to_lds)
#define HAVE_TDM 1
#else
#define HAVE_TDM 0
#endif

static __device__ __forceinline__ v8f wmma_bf16(v16bf a, v16bf b, v8f c) {
  // 8 args: (neg_a, A, neg_b, B, c_mod, C, reuse_a, reuse_b)
  return __builtin_amdgcn_wmma_f32_16x16x32_bf16(false, a, false, b, (short)0, c, false, false);
}

// ---- fragment loaders (16x16x32 bf16 operands, wave32; contiguous k = 16*(lane/16) + i in A and B) ----

// A fragment (16x32), row-major source: element(m,k) = src[m*lda + k]
// lanes 0-15: rows 0-15, K = 0..15; lanes 16-31: same rows, K = 16..31
static __device__ __forceinline__ v16bf load_a_rm(const bf16* src, int lda, int lane) {
  const int m  = lane & 15;
  const int kb = (lane >> 4) << 4; // 0 or 16
  const bf16* p = src + (size_t)m * lda + kb;
  v16bf a;
#pragma unroll
  for (int i = 0; i < 16; ++i) a[i] = p[i];
  return a;
}

// A fragment (16x32), column-major source: element(m,k) = src[k*ld + m]
static __device__ __forceinline__ v16bf load_a_cm(const bf16* src, int ld, int lane) {
  const int m  = lane & 15;
  const int kb = (lane >> 4) << 4;
  v16bf a;
#pragma unroll
  for (int i = 0; i < 16; ++i) a[i] = src[(size_t)(kb + i) * ld + m];
  return a;
}

// B fragment (32x16) from an N x K row-major matrix (implicit transpose):
// element(k,n) = src[n*ld + k].  lanes 0-15: K=0..15; lanes 16-31: K=16..31
static __device__ __forceinline__ v16bf load_b_nk(const bf16* src, int ld, int lane) {
  const int n  = lane & 15;
  const int kb = (lane >> 4) << 4; // 0 or 16
  const bf16* p = src + (size_t)n * ld + kb;
  v16bf b;
#pragma unroll
  for (int i = 0; i < 16; ++i) b[i] = p[i];
  return b;
}

// B fragment (32x16) from a K x N row-major matrix (e.g. P^T staged in LDS)
static __device__ __forceinline__ v16bf load_b_kn(const bf16* src, int ld, int lane) {
  const int n  = lane & 15;
  const int kb = (lane >> 4) << 4;
  v16bf b;
#pragma unroll
  for (int i = 0; i < 16; ++i) b[i] = src[(kb + i) * ld + n];
  return b;
}

// ---- Tensor Data Mover: async load of a 32x64 bf16 tile (row-major, ld=64) to LDS ----
// D# per CDNA5 ISA: group0 = {count|flags, lds_addr,
// global_addr[31:0], global_addr[56:32]|type=2}; group1 packs data_size=2B,
// tensor_dim0=64, tensor_dim1=32, tile_dim0=64, tile_dim1=32, dim0_stride=64.
static __device__ __forceinline__ void tdm_load_tile_32x64(const bf16* gsrc, bf16* ldsDst) {
#if HAVE_TDM
  const unsigned long long ga = (unsigned long long)(size_t)gsrc;
  const unsigned int lds = (unsigned int)(size_t)ldsDst; // generic low 32b == LDS byte addr
  v4u g0;
  g0[0] = 1u;                                                   // count=1, user descriptor
  g0[1] = lds;                                                  // lds_addr [63:32]
  g0[2] = (unsigned int)ga;                                     // global_addr [95:64]
  g0[3] = (unsigned int)((ga >> 32) & 0x01FFFFFFu) | (2u << 30); // addr[56:32] | type=2
  v8i g1;
  g1[0] = 0x00010000;        // workgroup_mask=0, data_size=1 -> 2-byte elements
  g1[1] = (int)(64u << 16);  // tensor_dim0 = 64   (bits [79:48], low half)
  g1[2] = (int)(32u << 16);  // tensor_dim1 = 32   (bits [111:80], low half)
  g1[3] = (int)(64u << 16);  // tile_dim0   = 64   (bits [127:112])
  g1[4] = 32;                // tile_dim1   = 32, tile_dim2 = 0
  g1[5] = 64;                // tensor_dim0_stride = 64 elements
  g1[6] = 0;                 // dim1 stride unused for 2D tile
  g1[7] = 0;
  const v4i z4 = {0, 0, 0, 0};
#if __clang_major__ >= 23
  const v8i z8 = {0, 0, 0, 0, 0, 0, 0, 0};
  __builtin_amdgcn_tensor_load_to_lds(g0, g1, z4, z4, z8, 0);
#else
  __builtin_amdgcn_tensor_load_to_lds(g0, g1, z4, z4, 0);
#endif
#else
  // Fallback (host pass / no-TDM toolchain): per-wave synchronous copy.
  for (int i = (int)(threadIdx.x & 31); i < 32 * DKK; i += 32) ldsDst[i] = gsrc[i];
#endif
}

static __device__ __forceinline__ void tdm_wait_le2() {
#if HAVE_TDM
  __builtin_amdgcn_s_wait_tensorcnt(2); // newest double-buffer pair stays in flight
#endif
}
static __device__ __forceinline__ void tdm_wait_0() {
#if HAVE_TDM
  __builtin_amdgcn_s_wait_tensorcnt(0);
#endif
}

// ---------------- kernel 1: f32 -> bf16 ----------------
__global__ void f32_to_bf16_kernel(const float* __restrict__ in, bf16* __restrict__ out, int n) {
  int i = blockIdx.x * blockDim.x + threadIdx.x;
  if (i < n) out[i] = (bf16)in[i];
}

// ---------------- kernel 2: QKV projection GEMM ----------------
// q/k/v[o] = sum_d x[s,d]*W[o,d]; write bf16 to (B,H,S,DK) scatter layout.
__global__ void __launch_bounds__(256)
qkv_gemm_kernel(const bf16* __restrict__ xb,
                const bf16* __restrict__ Wq, const bf16* __restrict__ Wk,
                const bf16* __restrict__ Wv,
                bf16* __restrict__ Qo, bf16* __restrict__ Ko, bf16* __restrict__ Vo) {
  const int lane = threadIdx.x & 31;
  const int wave = threadIdx.x >> 5;
  const int m0 = blockIdx.x * 16;
  const int n0 = (blockIdx.y * 8 + wave) * 16;
  const bf16* W   = (blockIdx.z == 0) ? Wq : (blockIdx.z == 1) ? Wk : Wv;
  bf16*       Dst = (blockIdx.z == 0) ? Qo : (blockIdx.z == 1) ? Ko : Vo;

  v8f acc = {};
#pragma unroll 4
  for (int k = 0; k < DD; k += 32) {
    v16bf a = load_a_rm(xb + (size_t)m0 * DD + k, DD, lane);
    v16bf b = load_b_nk(W + (size_t)n0 * DD + k, DD, lane);
    acc = wmma_bf16(a, b, acc);
  }
  const int n  = n0 + (lane & 15);
  const int h  = n >> 6, dk = n & 63;
  const int rb = m0 + ((lane >> 4) << 3);
#pragma unroll
  for (int r = 0; r < 8; ++r) {
    const int gr = rb + r;
    const int b_ = gr >> 11;      // /2048
    const int s_ = gr & 2047;
    Dst[(((size_t)(b_ * HH + h) * SS) + s_) * DKK + dk] = (bf16)acc[r];
  }
}

// ---------------- kernel 3: interleaved RoPE (in-place on bf16 Q or K) ----------------
__global__ void rope_kernel(bf16* __restrict__ t, const float* __restrict__ cosE,
                            const float* __restrict__ sinE, const int* __restrict__ pos,
                            int npairs) {
  int p = blockIdx.x * blockDim.x + threadIdx.x;
  if (p >= npairs) return;
  const int i2 = p * 2;
  const int dk = i2 & (DKK - 1);
  const int s  = (i2 >> 6) & (SS - 1);
  const int tp = pos[s];
  const float c0 = cosE[tp * DKK + dk],     s0 = sinE[tp * DKK + dk];
  const float c1 = cosE[tp * DKK + dk + 1], s1 = sinE[tp * DKK + dk + 1];
  const float x0 = (float)t[i2], x1 = (float)t[i2 + 1];
  t[i2]     = (bf16)(x0 * c0 - x1 * s0);
  t[i2 + 1] = (bf16)(x1 * c1 + x0 * s1);
}

// ---------------- kernel 4: flash attention ----------------
// One 16-query tile per wave. K/V 32x64 tiles are TDM double-buffered into LDS
// (TENSORcnt pipelining).  Computes S^T = K.Q^T so the softmax reduce is one
// shfl_xor(16) + in-lane ops, then O^T = V^T.P^T with the C->B layout transpose
// of P^T staged through per-wave LDS.
__global__ void __launch_bounds__(32 * AW)
flash_attn_kernel(const bf16* __restrict__ Q, const bf16* __restrict__ K,
                  const bf16* __restrict__ V, bf16* __restrict__ O) {
  __shared__ bf16 ksh[AW][2][32 * DKK]; // 4KB per buffer
  __shared__ bf16 vsh[AW][2][32 * DKK];
  __shared__ bf16 psh[AW][32 * 16];     // P^T staging (32 keys x 16 queries)
  const int lane = threadIdx.x & 31;
  const int wave = threadIdx.x >> 5;
  const int gw = blockIdx.x * AW + wave;
  const int qt = gw & 127;   // S/16 = 128 query tiles
  const int bh = gw >> 7;    // b*H + h in [0, 64)
  const int q0 = qt * 16;
  const bf16* Qp = Q + (size_t)bh * SS * DKK;
  const bf16* Kp = K + (size_t)bh * SS * DKK;
  const bf16* Vp = V + (size_t)bh * SS * DKK;

  // Q^T as B operands: B(d, i) = Q[(q0+i)*DK + d], inner-d split 0..31 / 32..63
  const v16bf bq0 = load_b_nk(Qp + (size_t)q0 * DKK + 0,  DKK, lane);
  const v16bf bq1 = load_b_nk(Qp + (size_t)q0 * DKK + 32, DKK, lane);

  float m_i = -3.0e38f, l_i = 0.f;
  v8f o0 = {}, o1 = {}, o2 = {}, o3 = {}; // O^T tiles (d 0-15,16-31,32-47,48-63)
  const int qi   = q0 + (lane & 15);
  const int krow = (lane >> 4) << 3;

  // prologue: fill buffer 0
  tdm_load_tile_32x64(Kp, &ksh[wave][0][0]);
  tdm_load_tile_32x64(Vp, &vsh[wave][0][0]);
  int buf = 0;

  for (int j0 = 0; j0 < q0 + 16; j0 += 32) {
    if (j0 + 32 < q0 + 16) { // issue next tile pair into the other buffer
      tdm_load_tile_32x64(Kp + (size_t)(j0 + 32) * DKK, &ksh[wave][buf ^ 1][0]);
      tdm_load_tile_32x64(Vp + (size_t)(j0 + 32) * DKK, &vsh[wave][buf ^ 1][0]);
      tdm_wait_le2();      // current buffer's pair is complete
    } else {
      tdm_wait_0();
    }
    const bf16* kb = &ksh[wave][buf][0];
    const bf16* vb = &vsh[wave][buf][0];

    // S^T (32 keys x 16 queries) as two 16x16 C tiles; inner = DK = 2x32
    v8f s0 = {}, s1 = {};
    s0 = wmma_bf16(load_a_rm(kb + 0,            DKK, lane), bq0, s0);
    s0 = wmma_bf16(load_a_rm(kb + 32,           DKK, lane), bq1, s0);
    s1 = wmma_bf16(load_a_rm(kb + 16 * DKK + 0,  DKK, lane), bq0, s1);
    s1 = wmma_bf16(load_a_rm(kb + 16 * DKK + 32, DKK, lane), bq1, s1);

    // scale, causal mask, per-query running max (query == lane column)
    float tmax = -3.0e38f;
#pragma unroll
    for (int r = 0; r < 8; ++r) {
      float v0 = s0[r] * 0.125f; if (j0 + krow + r > qi)      v0 = -3.0e38f;
      float v1 = s1[r] * 0.125f; if (j0 + 16 + krow + r > qi) v1 = -3.0e38f;
      s0[r] = v0; s1[r] = v1;
      tmax = fmaxf(tmax, fmaxf(v0, v1));
    }
    tmax = fmaxf(tmax, __shfl_xor(tmax, 16, 32));
    const float m_new = fmaxf(m_i, tmax);
    const float resc  = __expf(m_i - m_new);
    float tsum = 0.f;
#pragma unroll
    for (int r = 0; r < 8; ++r) {
      const float e0 = __expf(s0[r] - m_new);
      const float e1 = __expf(s1[r] - m_new);
      s0[r] = e0; s1[r] = e1;
      tsum += e0 + e1;
    }
    tsum += __shfl_xor(tsum, 16, 32);
    m_i = m_new;
    l_i = l_i * resc + tsum;
#pragma unroll
    for (int r = 0; r < 8; ++r) { o0[r] *= resc; o1[r] *= resc; o2[r] *= resc; o3[r] *= resc; }

    // P^T C-layout -> LDS row-major [key][query] -> reload as B fragment
    bf16* pb = &psh[wave][0];
#pragma unroll
    for (int r = 0; r < 8; ++r) {
      pb[(krow + r) * 16 + (lane & 15)]      = (bf16)s0[r];
      pb[(16 + krow + r) * 16 + (lane & 15)] = (bf16)s1[r];
    }
    __builtin_amdgcn_wave_barrier(); // LDS ops are in-order per wave; stop reordering
    const v16bf pfrag = load_b_kn(pb, 16, lane);

    // O^T += V^T . P^T : A(m=d, k=key) = vb[key*DK + d]
    o0 = wmma_bf16(load_a_cm(vb + 0,  DKK, lane), pfrag, o0);
    o1 = wmma_bf16(load_a_cm(vb + 16, DKK, lane), pfrag, o1);
    o2 = wmma_bf16(load_a_cm(vb + 32, DKK, lane), pfrag, o2);
    o3 = wmma_bf16(load_a_cm(vb + 48, DKK, lane), pfrag, o3);
    __builtin_amdgcn_wave_barrier();

    buf ^= 1;
  }

  // normalize and store O^T -> attn_out (B, S, D) bf16
  const float inv = 1.0f / l_i;
  const int b_ = bh >> 4, h_ = bh & 15;
  bf16* orow = O + ((size_t)(b_ * SS + q0 + (lane & 15))) * DD + h_ * DKK;
  const int dbase = (lane >> 4) << 3;
#pragma unroll
  for (int r = 0; r < 8; ++r) {
    orow[ 0 + dbase + r] = (bf16)(o0[r] * inv);
    orow[16 + dbase + r] = (bf16)(o1[r] * inv);
    orow[32 + dbase + r] = (bf16)(o2[r] * inv);
    orow[48 + dbase + r] = (bf16)(o3[r] * inv);
  }
}

// ---------------- kernel 5: output projection (fp32 result) ----------------
__global__ void __launch_bounds__(256)
out_proj_kernel(const bf16* __restrict__ A, const bf16* __restrict__ Wo,
                float* __restrict__ out) {
  const int lane = threadIdx.x & 31;
  const int wave = threadIdx.x >> 5;
  const int m0 = blockIdx.x * 16;
  const int n0 = (blockIdx.y * 8 + wave) * 16;
  v8f acc = {};
#pragma unroll 4
  for (int k = 0; k < DD; k += 32) {
    acc = wmma_bf16(load_a_rm(A + (size_t)m0 * DD + k, DD, lane),
                    load_b_nk(Wo + (size_t)n0 * DD + k, DD, lane), acc);
  }
  const int n  = n0 + (lane & 15);
  const int rb = m0 + ((lane >> 4) << 3);
#pragma unroll
  for (int r = 0; r < 8; ++r)
    out[(size_t)(rb + r) * DD + n] = acc[r];
}

// ---------------- launch ----------------
extern "C" void kernel_launch(void* const* d_in, const int* in_sizes, int n_in,
                              void* d_out, int out_size, void* d_ws, size_t ws_size,
                              hipStream_t stream) {
  (void)in_sizes; (void)n_in; (void)out_size; (void)ws_size;
  const float* x    = (const float*)d_in[0];
  const float* Wq   = (const float*)d_in[1];
  const float* Wk   = (const float*)d_in[2];
  const float* Wv   = (const float*)d_in[3];
  const float* Wo   = (const float*)d_in[4];
  const float* cosE = (const float*)d_in[5];
  const float* sinE = (const float*)d_in[6];
  const int*   pos  = (const int*)d_in[7];
  float* out = (float*)d_out;

  char* ws = (char*)d_ws;
  const size_t MiB = 1024 * 1024;
  bf16* xb  = (bf16*)(ws + 0);        // 16 MiB: x bf16 (8192x1024)
  bf16* wqb = (bf16*)(ws + 16 * MiB); // 2 MiB each
  bf16* wkb = (bf16*)(ws + 18 * MiB);
  bf16* wvb = (bf16*)(ws + 20 * MiB);
  bf16* wob = (bf16*)(ws + 22 * MiB);
  bf16* qb  = (bf16*)(ws + 24 * MiB); // 16 MiB (B,H,S,DK)
  bf16* kb  = (bf16*)(ws + 40 * MiB); // 16 MiB
  bf16* vb  = (bf16*)(ws + 56 * MiB); // 16 MiB
  bf16* ab  = (bf16*)(ws + 72 * MiB); // 16 MiB attn out (B,S,D)

  const int nx = BSR * DD;
  const int nw = DD * DD;
  f32_to_bf16_kernel<<<(nx + 255) / 256, 256, 0, stream>>>(x,  xb,  nx);
  f32_to_bf16_kernel<<<(nw + 255) / 256, 256, 0, stream>>>(Wq, wqb, nw);
  f32_to_bf16_kernel<<<(nw + 255) / 256, 256, 0, stream>>>(Wk, wkb, nw);
  f32_to_bf16_kernel<<<(nw + 255) / 256, 256, 0, stream>>>(Wv, wvb, nw);
  f32_to_bf16_kernel<<<(nw + 255) / 256, 256, 0, stream>>>(Wo, wob, nw);

  qkv_gemm_kernel<<<dim3(BSR / 16, DD / 16 / 8, 3), 256, 0, stream>>>(
      xb, wqb, wkb, wvb, qb, kb, vb);

  const int npairs = BSR * DD / 2;
  rope_kernel<<<(npairs + 255) / 256, 256, 0, stream>>>(qb, cosE, sinE, pos, npairs);
  rope_kernel<<<(npairs + 255) / 256, 256, 0, stream>>>(kb, cosE, sinE, pos, npairs);

  flash_attn_kernel<<<(BB * HH * (SS / 16)) / AW, 32 * AW, 0, stream>>>(qb, kb, vb, ab);

  out_proj_kernel<<<dim3(BSR / 16, DD / 16 / 8), 256, 0, stream>>>(ab, wob, out);
}
